// SwinTransformerBlock_69544110457626
// MI455X (gfx1250) — matrix operand order repaired and checked by reading, not compiled, not run on hardware
//
#include <hip/hip_runtime.h>
#include <cstdint>
#include <cstddef>

// ---------------------------------------------------------------------------
// Swin block for MI455X (gfx1250, wave32, WMMA 16x16x32 f16->f32,
// async global->LDS staging double-buffered in the GEMM)
// ---------------------------------------------------------------------------

typedef __attribute__((ext_vector_type(16))) _Float16 v16h;
typedef __attribute__((ext_vector_type(8)))  float    v8f;

union F16x16 { v16h v; _Float16 h[16]; uint4 u4[2]; };

#define BATCH   64
#define CDIM    192
#define HDIM    56
#define MWIN    7
#define SHIFT   3
#define NHEADS  6
#define DHEAD   32
#define HIDDIM  768
#define LTOK    49
#define NWIN    64
#define ROWS    (BATCH * NWIN * LTOK)   // 200704

// ---------------------------------------------------------------------------
// f32 -> f16 weight conversion
// ---------------------------------------------------------------------------
__global__ __launch_bounds__(256) void cvt_f32_f16_kernel(
    const float* __restrict__ src, _Float16* __restrict__ dst, int n)
{
    int i = blockIdx.x * 256 + threadIdx.x;
    if (i < n) dst[i] = (_Float16)src[i];
}

// ---------------------------------------------------------------------------
// NCHW -> window-partition layout [B*NWIN*L, C] (f32 residual stream)
// ---------------------------------------------------------------------------
__global__ __launch_bounds__(256) void win_partition_kernel(
    const float* __restrict__ x, float* __restrict__ res)
{
    size_t idx = (size_t)blockIdx.x * 256 + threadIdx.x;
    if (idx >= (size_t)ROWS * CDIM) return;
    int c   = (int)(idx % CDIM);
    size_t row = idx / CDIM;
    int l   = (int)(row % LTOK);
    int bw  = (int)(row / LTOK);
    int win = bw & (NWIN - 1);
    int b   = bw >> 6;
    int h   = (win >> 3) * MWIN + l / MWIN;
    int w   = (win & 7)  * MWIN + l % MWIN;
    res[idx] = x[(((size_t)b * CDIM + c) * HDIM + h) * HDIM + w];
}

__device__ __forceinline__ size_t row_of(int b, int h, int w)
{
    int win = (h / MWIN) * 8 + (w / MWIN);
    int l   = (h % MWIN) * MWIN + (w % MWIN);
    return ((size_t)b * NWIN + win) * LTOK + l;
}

// window layout -> window layout, spatial roll by (-SHIFT,-SHIFT)
__global__ __launch_bounds__(256) void shift_win_kernel(
    const float* __restrict__ src, float* __restrict__ dst)
{
    size_t idx = (size_t)blockIdx.x * 256 + threadIdx.x;
    if (idx >= (size_t)ROWS * CDIM) return;
    int c   = (int)(idx % CDIM);
    size_t row = idx / CDIM;
    int l   = (int)(row % LTOK);
    int bw  = (int)(row / LTOK);
    int win = bw & (NWIN - 1);
    int b   = bw >> 6;
    int h   = (win >> 3) * MWIN + l / MWIN;
    int w   = (win & 7)  * MWIN + l % MWIN;
    int hs = h + SHIFT; if (hs >= HDIM) hs -= HDIM;
    int ws = w + SHIFT; if (ws >= HDIM) ws -= HDIM;
    dst[idx] = src[row_of(b, hs, ws) * CDIM + c];
}

// window layout -> NCHW with spatial roll by (+SHIFT,+SHIFT)
__global__ __launch_bounds__(256) void win_scatter_kernel(
    const float* __restrict__ res, float* __restrict__ out)
{
    size_t idx = (size_t)blockIdx.x * 256 + threadIdx.x;
    if (idx >= (size_t)BATCH * CDIM * HDIM * HDIM) return;
    int w = (int)(idx % HDIM); size_t t = idx / HDIM;
    int h = (int)(t % HDIM);   t /= HDIM;
    int c = (int)(t % CDIM);
    int b = (int)(t / CDIM);
    int hs = h - SHIFT; if (hs < 0) hs += HDIM;
    int ws = w - SHIFT; if (ws < 0) ws += HDIM;
    out[idx] = res[row_of(b, hs, ws) * CDIM + c];
}

// ---------------------------------------------------------------------------
// LayerNorm: one wave per token row (C=192 -> 6 elems/lane), f32 in, f16 out
// ---------------------------------------------------------------------------
__global__ __launch_bounds__(256) void ln_f16_kernel(
    const float* __restrict__ in, const float* __restrict__ g,
    const float* __restrict__ b, _Float16* __restrict__ out)
{
    int lane = threadIdx.x & 31;
    size_t row = (size_t)blockIdx.x * 8 + (threadIdx.x >> 5);
    const float* p = in + row * CDIM;
    float v[6];
    float s = 0.f;
#pragma unroll
    for (int i = 0; i < 6; ++i) { v[i] = p[lane + i * 32]; s += v[i]; }
#pragma unroll
    for (int d = 16; d >= 1; d >>= 1) s += __shfl_xor(s, d, 32);
    float mu = s * (1.0f / CDIM);
    float s2 = 0.f;
#pragma unroll
    for (int i = 0; i < 6; ++i) { float t = v[i] - mu; s2 += t * t; }
#pragma unroll
    for (int d = 16; d >= 1; d >>= 1) s2 += __shfl_xor(s2, d, 32);
    float rstd = rsqrtf(s2 * (1.0f / CDIM) + 1e-5f);
    _Float16* q = out + row * CDIM;
#pragma unroll
    for (int i = 0; i < 6; ++i) {
        int c = lane + i * 32;
        q[c] = (_Float16)((v[i] - mu) * rstd * g[c] + b[c]);
    }
}

// ---------------------------------------------------------------------------
// WMMA GEMM: C[ROWS,N] = A[ROWS,K](f16) @ W[K,N](f16) + bias, fused epilogues.
// Workgroup tile 128(M) x 64(N); 8 waves, each wave 1 M-tile x 4 N-tiles.
// A tile staged via GLOBAL_LOAD_ASYNC_TO_LDS_B128 (ASYNCcnt), double-buffered;
// B tile prefetched into VGPRs one step ahead and transposed through LDS.
// ---------------------------------------------------------------------------
enum { EPI_BIAS_F16 = 0, EPI_RES_F32 = 1, EPI_GELU_F16 = 2 };

template <int EPI>
__global__ __launch_bounds__(256) void gemm_wmma_kernel(
    const _Float16* __restrict__ A, const _Float16* __restrict__ Bw,
    const float* __restrict__ bias, float* __restrict__ resio,
    _Float16* __restrict__ outh, int K, int N)
{
    __shared__ __align__(16) _Float16 sA[2][128 * 32];   // 2 x 8 KB
    __shared__ __align__(16) _Float16 sBt[2][64 * 32];   // 2 x 4 KB ([n][k])
    const int t    = threadIdx.x;
    const int lane = t & 31;
    const int wave = t >> 5;
    const int blockM = blockIdx.x * 128;
    const int blockN = blockIdx.y * 64;
    const int nl  = lane & 15;
    const int kb  = (lane >> 4) << 3;   // A fragment K base (groups of 8, +16 jump)
    const int kb2 = (lane >> 4) << 4;   // contiguous-16 K base; unused (B fragment uses kb)

    const int rS = t >> 2;              // staging row (A: 0..63, B: n 0..63)
    const int cS = (t & 3) << 3;        // staging col (8 halves = 16 B)

    // async copy of one 128x32 A tile into sA[buf]: 2 x b128 per lane
    auto issue_asyncA = [&](int k0, int buf) {
        const _Float16* src = A + (size_t)(blockM + rS) * K + k0 + cS;
        uint32_t d0 = (uint32_t)(uintptr_t)&sA[buf][rS * 32 + cS];
        uint32_t d1 = d0 + 64 * 32 * 2;
        uint64_t g0 = (uint64_t)(uintptr_t)src;
        uint64_t g1 = (uint64_t)(uintptr_t)(src + (size_t)64 * K);
        asm volatile("global_load_async_to_lds_b128 %0, %1, off"
                     :: "v"(d0), "v"(g0) : "memory");
        asm volatile("global_load_async_to_lds_b128 %0, %1, off"
                     :: "v"(d1), "v"(g1) : "memory");
    };

    v8f acc[4] = {};
    const int T = K >> 5;
    _Float16 bcur[8], bnxt[8];

    // prologue: tile 0 in flight
    issue_asyncA(0, 0);
    {
        const _Float16* src = Bw + (size_t)cS * N + blockN + rS;
#pragma unroll
        for (int j = 0; j < 8; ++j) bcur[j] = src[(size_t)j * N];
    }

    for (int ti = 0; ti < T; ++ti) {
        const int buf = ti & 1;
        const bool more = (ti + 1 < T);
        if (more) {
            issue_asyncA((ti + 1) << 5, buf ^ 1);           // overlap with compute
            const _Float16* src = Bw + (size_t)(((ti + 1) << 5) + cS) * N + blockN + rS;
#pragma unroll
            for (int j = 0; j < 8; ++j) bnxt[j] = src[(size_t)j * N];
        }
        {   // transpose-store current B tile: sBt[n][k]
            _Float16* dst = &sBt[buf][rS * 32 + cS];
#pragma unroll
            for (int j = 0; j < 8; ++j) dst[j] = bcur[j];
        }
        // wave-local: current tile's 2 async ops done (next tile may stay in flight)
        if (more) asm volatile("s_wait_asynccnt 0x2" ::: "memory");
        else      asm volatile("s_wait_asynccnt 0x0" ::: "memory");
        __syncthreads();    // all waves' async slices + Bt stores visible

        F16x16 af;
        {
            int m = wave * 16 + nl;
            af.u4[0] = *(const uint4*)&sA[buf][m * 32 + kb];
            af.u4[1] = *(const uint4*)&sA[buf][m * 32 + kb + 16];
        }
#pragma unroll
        for (int j = 0; j < 4; ++j) {
            F16x16 bf;
            int n = j * 16 + nl;
            bf.u4[0] = *(const uint4*)&sBt[buf][n * 32 + kb];
            bf.u4[1] = *(const uint4*)&sBt[buf][n * 32 + kb + 16];
            acc[j] = __builtin_amdgcn_wmma_f32_16x16x32_f16(
                false, af.v, false, bf.v, (short)0, acc[j], false, false);
        }
        __syncthreads();    // protect buf from being overwritten next-next tile
        if (more) {
#pragma unroll
            for (int j = 0; j < 8; ++j) bcur[j] = bnxt[j];
        }
    }

    const int hi = (lane >> 4) << 3;    // D layout: M = e + (lane<16 ? 0 : 8)
#pragma unroll
    for (int j = 0; j < 4; ++j) {
        int col = blockN + j * 16 + nl;
        float bb = bias[col];
#pragma unroll
        for (int e = 0; e < 8; ++e) {
            int row = blockM + wave * 16 + hi + e;
            float v = acc[j][e] + bb;
            if constexpr (EPI == EPI_BIAS_F16) {
                outh[(size_t)row * N + col] = (_Float16)v;
            } else if constexpr (EPI == EPI_RES_F32) {
                size_t idx = (size_t)row * N + col;
                resio[idx] = resio[idx] + v;
            } else {
                float gl = 0.5f * v * (1.0f + erff(v * 0.70710678118f));
                outh[(size_t)row * N + col] = (_Float16)gl;
            }
        }
    }
}

// ---------------------------------------------------------------------------
// Windowed attention: one wave per (window, head). d=32, L=49 padded to 64.
// scores = q@k^T (16 WMMA), bias+mask+softmax in regs, probs via LDS, a@v (16).
// ---------------------------------------------------------------------------
__global__ __launch_bounds__(128) void attn_wmma_kernel(
    const _Float16* __restrict__ qkv, _Float16* __restrict__ o,
    const float* __restrict__ biastab, const int* __restrict__ relidx,
    const unsigned char* __restrict__ maskp)
{
    __shared__ __align__(16) _Float16 sa[4][64 * 64];   // 32 KB: probs per wave
    const int lane  = threadIdx.x & 31;
    const int wslot = threadIdx.x >> 5;
    const int wid   = blockIdx.x * 4 + wslot;   // 0 .. 4096*6-1
    const int w     = wid / NHEADS;
    const int head  = wid % NHEADS;
    const int win   = w & (NWIN - 1);
    const int nl  = lane & 15;
    const int kb  = (lane >> 4) << 3;
    const int kb2 = (lane >> 4) << 4;
    const int hi  = (lane >> 4) << 3;

    const _Float16* qp = qkv + (size_t)w * LTOK * 576 + head * DHEAD;
    const _Float16* kp = qp + CDIM;       // +192
    const _Float16* vp = qp + 2 * CDIM;   // +384

    // q fragments (A layout) and k fragments (B layout: B[d][kl] = k[kl][d])
    F16x16 qf[4], kf[4];
#pragma unroll
    for (int i = 0; i < 4; ++i) {
        int m = i * 16 + nl;
        if (m < LTOK) {
            const _Float16* p = qp + (size_t)m * 576;
            qf[i].u4[0] = *(const uint4*)(p + kb);
            qf[i].u4[1] = *(const uint4*)(p + kb + 16);
            const _Float16* pk = kp + (size_t)m * 576;
            kf[i].u4[0] = *(const uint4*)(pk + kb);
            kf[i].u4[1] = *(const uint4*)(pk + kb + 16);
        } else {
            qf[i].u4[0] = make_uint4(0, 0, 0, 0); qf[i].u4[1] = make_uint4(0, 0, 0, 0);
            kf[i].u4[0] = make_uint4(0, 0, 0, 0); kf[i].u4[1] = make_uint4(0, 0, 0, 0);
        }
    }

    v8f sc[4][4] = {};
#pragma unroll
    for (int i = 0; i < 4; ++i)
#pragma unroll
        for (int j = 0; j < 4; ++j)
            sc[i][j] = __builtin_amdgcn_wmma_f32_16x16x32_f16(
                false, qf[i].v, false, kf[j].v, (short)0, sc[i][j], false, false);

    // bias + mask + softmax (rows live on 16-lane halves; width-16 reductions)
    const float scale = 0.17677669529663687f;   // 1/sqrt(32)
#pragma unroll
    for (int i = 0; i < 4; ++i) {
#pragma unroll
        for (int e = 0; e < 8; ++e) {
            int m = i * 16 + hi + e;
            float vals[4];
#pragma unroll
            for (int j = 0; j < 4; ++j) {
                int n = j * 16 + nl;
                float s = -1e9f;
                if (m < LTOK && n < LTOK) {
                    s = sc[i][j][e] * scale +
                        biastab[relidx[m * LTOK + n] * NHEADS + head];
                    if (maskp && maskp[(size_t)win * LTOK * LTOK + m * LTOK + n])
                        s = -1e9f;
                }
                vals[j] = s;
            }
            float rmax = fmaxf(fmaxf(vals[0], vals[1]), fmaxf(vals[2], vals[3]));
#pragma unroll
            for (int d = 8; d >= 1; d >>= 1) rmax = fmaxf(rmax, __shfl_xor(rmax, d, 16));
            float rsum = 0.f;
#pragma unroll
            for (int j = 0; j < 4; ++j) { vals[j] = __expf(vals[j] - rmax); rsum += vals[j]; }
#pragma unroll
            for (int d = 8; d >= 1; d >>= 1) rsum += __shfl_xor(rsum, d, 16);
            float inv = 1.0f / rsum;
#pragma unroll
            for (int j = 0; j < 4; ++j)
                sa[wslot][m * 64 + j * 16 + nl] = (_Float16)(vals[j] * inv);
        }
    }

    // o = a @ v : A = probs (64x64, from LDS), B = v (64x32), two K steps
    v8f oc[4][2] = {};
#pragma unroll
    for (int kk = 0; kk < 2; ++kk) {
        F16x16 vf[2];
#pragma unroll
        for (int j = 0; j < 2; ++j) {
#pragma unroll
            for (int e = 0; e < 16; ++e) {
                int kl = kk * 32 + kb + e + (e >= 8 ? 8 : 0);
                _Float16 val = (_Float16)0.f;
                if (kl < LTOK) val = vp[(size_t)kl * 576 + j * 16 + nl];
                vf[j].h[e] = val;
            }
        }
#pragma unroll
        for (int i = 0; i < 4; ++i) {
            F16x16 afg;
            int m = i * 16 + nl;
            const _Float16* pa = &sa[wslot][m * 64 + kk * 32 + kb];
            afg.u4[0] = *(const uint4*)pa;
            afg.u4[1] = *(const uint4*)(pa + 16);
#pragma unroll
            for (int j = 0; j < 2; ++j)
                oc[i][j] = __builtin_amdgcn_wmma_f32_16x16x32_f16(
                    false, afg.v, false, vf[j].v, (short)0, oc[i][j], false, false);
        }
    }

    _Float16* op = o + (size_t)w * LTOK * CDIM + head * DHEAD;
#pragma unroll
    for (int i = 0; i < 4; ++i)
#pragma unroll
        for (int e = 0; e < 8; ++e) {
            int m = i * 16 + hi + e;
            if (m < LTOK) {
#pragma unroll
                for (int j = 0; j < 2; ++j)
                    op[(size_t)m * CDIM + j * 16 + nl] = (_Float16)oc[i][j][e];
            }
        }
}

// ---------------------------------------------------------------------------
// Host orchestration
// ---------------------------------------------------------------------------
extern "C" void kernel_launch(void* const* d_in, const int* in_sizes, int n_in,
                              void* d_out, int out_size, void* d_ws, size_t ws_size,
                              hipStream_t stream)
{
    (void)in_sizes; (void)n_in; (void)out_size; (void)ws_size;
    const float* x = (const float*)d_in[0];
    // setup_inputs dict insertion order: x, params{w: Wqkv,bqkv,Wo,bo,bias,g1,b1,
    // g2,b2,W1,b1m,W2,b2m; sw: same}, rel_idx, attn_mask
    auto P = [&](int layer, int k) -> const float* {
        return (const float*)d_in[1 + layer * 13 + k];
    };
    const int* relidx = (const int*)d_in[27];
    const unsigned char* mask = (const unsigned char*)d_in[28];
    float* out = (float*)d_out;

    // workspace layout (~664 MB total)
    char* ws = (char*)d_ws;
    float*    resA = (float*)(ws);                          // 154,140,672 B
    float*    resB = (float*)(ws + 154140672ULL);           // 154,140,672 B
    _Float16* xln  = (_Float16*)(ws + 308281344ULL);        //  77,070,336 B
    _Float16* qkvh = (_Float16*)(ws + 385351680ULL);        // 231,211,008 B
    _Float16* oh   = (_Float16*)(ws + 616562688ULL);        //  77,070,336 B
    _Float16* hh   = (_Float16*)(ws + 385351680ULL);        // reuse qkv+o
    _Float16* w16  = (_Float16*)(ws + 693633024ULL);        // 2 x 884,736 B

    auto conv = [&](const float* src, _Float16* dst, int n) {
        cvt_f32_f16_kernel<<<(n + 255) / 256, 256, 0, stream>>>(src, dst, n);
    };
    for (int l = 0; l < 2; ++l) {
        _Float16* base = w16 + (size_t)l * 442368;
        conv(P(l, 0),  base,           CDIM * 3 * CDIM);   // Wqkv 110592
        conv(P(l, 2),  base + 110592,  CDIM * CDIM);       // Wo   36864
        conv(P(l, 9),  base + 147456,  CDIM * HIDDIM);     // W1   147456
        conv(P(l, 11), base + 294912,  HIDDIM * CDIM);     // W2   147456
    }

    const int ELT_BLOCKS = (ROWS * CDIM + 255) / 256;      // 150528
    win_partition_kernel<<<ELT_BLOCKS, 256, 0, stream>>>(x, resA);

    auto layer = [&](float* res, int l, const unsigned char* mk) {
        _Float16* base = w16 + (size_t)l * 442368;
        ln_f16_kernel<<<ROWS / 8, 256, 0, stream>>>(res, P(l, 5), P(l, 6), xln);
        gemm_wmma_kernel<EPI_BIAS_F16><<<dim3(ROWS / 128, 9), 256, 0, stream>>>(
            xln, base, P(l, 1), nullptr, qkvh, CDIM, 3 * CDIM);
        attn_wmma_kernel<<<(BATCH * NWIN * NHEADS) / 4, 128, 0, stream>>>(
            qkvh, oh, P(l, 4), relidx, mk);
        gemm_wmma_kernel<EPI_RES_F32><<<dim3(ROWS / 128, 3), 256, 0, stream>>>(
            oh, base + 110592, P(l, 3), res, nullptr, CDIM, CDIM);
        ln_f16_kernel<<<ROWS / 8, 256, 0, stream>>>(res, P(l, 7), P(l, 8), xln);
        gemm_wmma_kernel<EPI_GELU_F16><<<dim3(ROWS / 128, 12), 256, 0, stream>>>(
            xln, base + 147456, P(l, 10), nullptr, hh, CDIM, HIDDIM);
        gemm_wmma_kernel<EPI_RES_F32><<<dim3(ROWS / 128, 3), 256, 0, stream>>>(
            hh, base + 294912, P(l, 12), res, nullptr, HIDDIM, CDIM);
    };

    layer(resA, 0, nullptr);
    shift_win_kernel<<<ELT_BLOCKS, 256, 0, stream>>>(resA, resB);
    layer(resB, 1, mask);
    win_scatter_kernel<<<ELT_BLOCKS, 256, 0, stream>>>(resB, out);
}
